// NodeNetwork_29884382446409
// MI455X (gfx1250) — matrix operand order repaired and checked by reading, not compiled, not run on hardware
//
#include <hip/hip_runtime.h>

// ---------------------------------------------------------------------------
// NodeNetwork: agg = sum(message, deg); x = [agg|feat|gfeat];
// h1 = relu(x@W1+b1); h2 = relu(h1@W2+b2); out = h2@W3+b3; out /= (||out||+1e-8)
//
// MI455X (gfx1250, wave32) strategy:
//   * HBM-bound: ~4.9 GB @ 23.3 TB/s => ~210 us floor. Stream `message` once
//     with 128-bit non-temporal loads (read-once, keep L2 for weights).
//   * All GEMMs on v_wmma_f32_16x16x32_bf16 (bf16 operands, f32 accumulate):
//     ~198 GFLOP rides the XDL path and stays under the memory roofline.
//   * 1 workgroup (8 waves) per 16-node tile; weights pre-packed into the ISA
//     B-fragment layout in d_ws so each lane loads 32 contiguous bytes (L2-hot).
// ---------------------------------------------------------------------------

typedef __attribute__((ext_vector_type(16))) __bf16 v16bf;
typedef __attribute__((ext_vector_type(8)))  float  v8f;
typedef __attribute__((ext_vector_type(4)))  float  f4;

#define NN      500000
#define DEG     16
#define FDIM    128
#define K1      384     // 3*FDIM
#define H1DIM   256
#define H2DIM   256
#define ODIM    128

#define XSTRIDE 392     // 384 + 8 pad (bf16 units) -> conflict-free ds_load_b128
#define HSTRIDE 264     // 256 + 8 pad
#define OSTRIDE 132     // 128 + 4 pad (f32 units)

// fragment counts (bf16 elements) in d_ws
#define W1FRAG  (K1    * H1DIM)   // 98304
#define W2FRAG  (H1DIM * H2DIM)   // 65536
#define W3FRAG  (H2DIM * ODIM)    // 32768

__device__ __forceinline__ unsigned short f32_to_bf16(float f) {
  union { float f; unsigned int u; } x; x.f = f;
  unsigned int u = x.u;
  u += 0x7FFFu + ((u >> 16) & 1u);      // round-to-nearest-even
  return (unsigned short)(u >> 16);
}

// Pack 4 floats -> 4 bf16, single 8-byte LDS store.
__device__ __forceinline__ void store_bf16x4(unsigned short* p, f4 v) {
  union { unsigned short s[4]; uint2 u; } q;
  q.s[0] = f32_to_bf16(v.x); q.s[1] = f32_to_bf16(v.y);
  q.s[2] = f32_to_bf16(v.z); q.s[3] = f32_to_bf16(v.w);
  *(uint2*)p = q.u;
}

// A-fragment gather from LDS activation tile (row-major bf16, padded stride).
// ISA 16-bit A layout: lanes 0-15 -> K {0..7,16..23}; lanes 16-31 -> K {8..15,24..31}.
__device__ __forceinline__ v16bf load_a_frag(const unsigned short* lds_row,
                                             int kbase, int grp) {
  union { int4 q[2]; v16bf v; } u;
  u.q[0] = *(const int4*)(lds_row + kbase + (grp ? 8  : 0));
  u.q[1] = *(const int4*)(lds_row + kbase + (grp ? 24 : 16));
  return u.v;
}

// B-fragment: pre-packed, lane-contiguous 32B in d_ws (default TH -> stays in L2).
__device__ __forceinline__ v16bf load_b_frag(const unsigned short* wf,
                                             int frag, int lane) {
  const int4* p = (const int4*)(wf + (((size_t)frag * 32) + lane) * 16);
  union { int4 q[2]; v16bf v; } u;
  u.q[0] = p[0];
  u.q[1] = p[1];
  return u.v;
}

// Pack W[K][Nout] (f32, row-major) into bf16 WMMA B-fragments:
// dst[((kt*nct+ct)*32 + lane)*16 + e] = bf16(W[kt*32 + (lane&16?8:0) + (e&7) + (e&8?16:0)][ct*16 + lane&15])
__global__ __launch_bounds__(256) void pack_weights_kernel(
    const float* __restrict__ W, unsigned short* __restrict__ dst,
    int K, int Nout) {
  int idx = blockIdx.x * blockDim.x + threadIdx.x;
  if (idx >= K * Nout) return;
  int nct  = Nout >> 4;
  int e    = idx & 15;
  int lane = (idx >> 4) & 31;
  int frag = idx >> 9;
  int ct   = frag % nct;
  int kt   = frag / nct;
  int k = kt * 32 + ((lane & 16) ? 8 : 0) + (e & 7) + ((e & 8) ? 16 : 0);
  int n = ct * 16 + (lane & 15);
  dst[idx] = f32_to_bf16(W[(size_t)k * Nout + n]);
}

__global__ __launch_bounds__(256) void node_mlp_kernel(
    const float* __restrict__ message,
    const float* __restrict__ features,
    const float* __restrict__ gfeat,
    const unsigned short* __restrict__ w1f, const float* __restrict__ b1,
    const unsigned short* __restrict__ w2f, const float* __restrict__ b2,
    const unsigned short* __restrict__ w3f, const float* __restrict__ b3,
    float* __restrict__ out) {
  __shared__ __align__(16) unsigned short xs[16 * XSTRIDE];
  __shared__ __align__(16) unsigned short hs[16 * HSTRIDE];
  __shared__ __align__(16) float outs[16 * OSTRIDE];
  __shared__ float invn[16];

  const int t     = threadIdx.x;
  const int lane  = t & 31;
  const int wave  = t >> 5;
  const int grp   = lane >> 4;        // A-fragment K-group
  const int ln    = lane & 15;        // N (col) within a 16-wide tile
  const int node0 = blockIdx.x * 16;

  // ---- Phase 0: 128-bit non-temporal streaming aggregation -> bf16 LDS tile
  // 512 float4 tasks (16 rows x 32 col-groups), 2 per thread.
  for (int i = 0; i < 2; ++i) {
    int task = t + 256 * i;
    int r = task >> 5, cg = task & 31;      // cg = float4 column group
    const f4* mp = (const f4*)(message + (size_t)(node0 + r) * (DEG * FDIM)) + cg;
    f4 s = {0.f, 0.f, 0.f, 0.f};
#pragma unroll
    for (int d = 0; d < DEG; ++d) {
      f4 v = __builtin_nontemporal_load(mp + d * (FDIM / 4));
      s += v;
    }
    store_bf16x4(xs + r * XSTRIDE + cg * 4, s);
    f4 fv = __builtin_nontemporal_load(
        (const f4*)(features + (size_t)(node0 + r) * FDIM) + cg);
    store_bf16x4(xs + r * XSTRIDE + 128 + cg * 4, fv);
    f4 gv = __builtin_nontemporal_load(
        (const f4*)(gfeat + (size_t)(node0 + r) * FDIM) + cg);
    store_bf16x4(xs + r * XSTRIDE + 256 + cg * 4, gv);
  }
  __syncthreads();

  // ---- Layer 1: x[16x384] @ W1[384x256] -> relu -> hs -----------------------
  {
    v8f acc0 = {}; v8f acc1 = {};
    const int ct0 = wave * 2, ct1 = wave * 2 + 1;
    const unsigned short* arow = xs + ln * XSTRIDE;
    for (int kt = 0; kt < K1 / 32; ++kt) {
      v16bf a  = load_a_frag(arow, kt * 32, grp);
      v16bf b0 = load_b_frag(w1f, kt * 16 + ct0, lane);
      v16bf bb = load_b_frag(w1f, kt * 16 + ct1, lane);
      acc0 = __builtin_amdgcn_wmma_f32_16x16x32_bf16(false, a, false, b0,
                                                     (short)0, acc0, false, false);
      acc1 = __builtin_amdgcn_wmma_f32_16x16x32_bf16(false, a, false, bb,
                                                     (short)0, acc1, false, false);
    }
    float bv0 = b1[ct0 * 16 + ln];
    float bv1 = b1[ct1 * 16 + ln];
#pragma unroll
    for (int v = 0; v < 8; ++v) {      // C/D layout: lane=N, vgpr=M (+8 hi lanes)
      int m = v + 8 * grp;
      float f0 = acc0[v] + bv0; f0 = f0 > 0.f ? f0 : 0.f;
      float f1 = acc1[v] + bv1; f1 = f1 > 0.f ? f1 : 0.f;
      hs[m * HSTRIDE + ct0 * 16 + ln] = f32_to_bf16(f0);
      hs[m * HSTRIDE + ct1 * 16 + ln] = f32_to_bf16(f1);
    }
  }
  __syncthreads();

  // ---- Layer 2: h1[16x256] @ W2[256x256] -> relu -> hs (in place) -----------
  {
    v8f acc0 = {}; v8f acc1 = {};
    const int ct0 = wave * 2, ct1 = wave * 2 + 1;
    const unsigned short* arow = hs + ln * HSTRIDE;
    for (int kt = 0; kt < H1DIM / 32; ++kt) {
      v16bf a  = load_a_frag(arow, kt * 32, grp);
      v16bf b0 = load_b_frag(w2f, kt * 16 + ct0, lane);
      v16bf bb = load_b_frag(w2f, kt * 16 + ct1, lane);
      acc0 = __builtin_amdgcn_wmma_f32_16x16x32_bf16(false, a, false, b0,
                                                     (short)0, acc0, false, false);
      acc1 = __builtin_amdgcn_wmma_f32_16x16x32_bf16(false, a, false, bb,
                                                     (short)0, acc1, false, false);
    }
    __syncthreads();                   // all reads of h1 done before overwrite
    float bv0 = b2[ct0 * 16 + ln];
    float bv1 = b2[ct1 * 16 + ln];
#pragma unroll
    for (int v = 0; v < 8; ++v) {
      int m = v + 8 * grp;
      float f0 = acc0[v] + bv0; f0 = f0 > 0.f ? f0 : 0.f;
      float f1 = acc1[v] + bv1; f1 = f1 > 0.f ? f1 : 0.f;
      hs[m * HSTRIDE + ct0 * 16 + ln] = f32_to_bf16(f0);
      hs[m * HSTRIDE + ct1 * 16 + ln] = f32_to_bf16(f1);
    }
  }
  __syncthreads();

  // ---- Layer 3: h2[16x256] @ W3[256x128] -> outs (f32) ----------------------
  {
    v8f acc = {};
    const int ct = wave;               // 8 col tiles, one per wave
    const unsigned short* arow = hs + ln * HSTRIDE;
    for (int kt = 0; kt < H2DIM / 32; ++kt) {
      v16bf a = load_a_frag(arow, kt * 32, grp);
      v16bf b = load_b_frag(w3f, kt * 8 + ct, lane);
      acc = __builtin_amdgcn_wmma_f32_16x16x32_bf16(false, a, false, b,
                                                    (short)0, acc, false, false);
    }
    float bv = b3[ct * 16 + ln];
#pragma unroll
    for (int v = 0; v < 8; ++v) {
      int m = v + 8 * grp;
      outs[m * OSTRIDE + ct * 16 + ln] = acc[v] + bv;
    }
  }
  __syncthreads();

  // ---- Row L2 norms (16 lanes, padded stride -> conflict-free) --------------
  if (t < 16) {
    const float* row = outs + t * OSTRIDE;
    float s = 0.f;
    for (int cg = 0; cg < ODIM / 4; ++cg) {
      f4 v = *(const f4*)(row + cg * 4);
      s += v.x * v.x + v.y * v.y + v.z * v.z + v.w * v.w;
    }
    invn[t] = 1.0f / (sqrtf(s) + 1e-8f);
  }
  __syncthreads();

  // ---- Normalized 128-bit non-temporal writeout -----------------------------
  for (int i = 0; i < 2; ++i) {
    int task = t + 256 * i;
    int r = task >> 5, cg = task & 31;
    f4 v = *(const f4*)(outs + r * OSTRIDE + cg * 4);
    v *= invn[r];
    __builtin_nontemporal_store(v, (f4*)(out + (size_t)(node0 + r) * ODIM) + cg);
  }
}

extern "C" void kernel_launch(void* const* d_in, const int* in_sizes, int n_in,
                              void* d_out, int out_size, void* d_ws, size_t ws_size,
                              hipStream_t stream) {
  (void)in_sizes; (void)n_in; (void)out_size; (void)ws_size;
  const float* message  = (const float*)d_in[0];
  const float* features = (const float*)d_in[1];
  const float* gfeat    = (const float*)d_in[2];
  const float* W1 = (const float*)d_in[3];
  const float* b1 = (const float*)d_in[4];
  const float* W2 = (const float*)d_in[5];
  const float* b2 = (const float*)d_in[6];
  const float* W3 = (const float*)d_in[7];
  const float* b3 = (const float*)d_in[8];
  float* out = (float*)d_out;

  unsigned short* w1f = (unsigned short*)d_ws;
  unsigned short* w2f = w1f + W1FRAG;
  unsigned short* w3f = w2f + W2FRAG;

  // Pack weights into WMMA B-fragment layout (deterministic, every call).
  pack_weights_kernel<<<(W1FRAG + 255) / 256, 256, 0, stream>>>(W1, w1f, K1, H1DIM);
  pack_weights_kernel<<<(W2FRAG + 255) / 256, 256, 0, stream>>>(W2, w2f, H1DIM, H2DIM);
  pack_weights_kernel<<<(W3FRAG + 255) / 256, 256, 0, stream>>>(W3, w3f, H2DIM, ODIM);

  node_mlp_kernel<<<NN / 16, 256, 0, stream>>>(
      message, features, gfeat, w1f, b1, w2f, b2, w3f, b3, out);
}
